// DeepGpcmCornModel_27298812133852
// MI455X (gfx1250) — matrix operand order repaired and checked by reading, not compiled, not run on hardware
//
#include <hip/hip_runtime.h>

// ---------------------------------------------------------------------------
// CDNA5 (gfx1250) fused DKVMN/CORN pipeline.
// WMMA: v_wmma_f32_16x16x32_bf16 (wave32, A 16x32 / B 32x16 / C 16x16 f32).
// TDM:  tensor_load_to_lds (guarded; 5-arg clang-22 / 6-arg clang-23 forms).
// ---------------------------------------------------------------------------

typedef __attribute__((ext_vector_type(16))) __bf16 v16bf;
typedef __attribute__((ext_vector_type(8)))  float  v8f;
typedef __attribute__((ext_vector_type(4)))  unsigned int v4u;
typedef __attribute__((ext_vector_type(4)))  int v4i;
typedef __attribute__((ext_vector_type(8)))  int v8i;

union Frag16 { v16bf v; unsigned int u[8]; };

__device__ __forceinline__ unsigned short f2bf(float x) {
  unsigned int u = __float_as_uint(x);
  u += 0x7FFFu + ((u >> 16) & 1u);   // round-to-nearest-even
  return (unsigned short)(u >> 16);
}
__device__ __forceinline__ float sigmoidf_(float x) {
  return 1.0f / (1.0f + __expf(-x));
}

#define NBT 262144L   // B*T
#define NQn 1000

#if __has_builtin(__builtin_amdgcn_tensor_load_to_lds) && \
    __has_builtin(__builtin_amdgcn_s_wait_tensorcnt)
#define HAVE_TDM 1
#else
#define HAVE_TDM 0
#endif

#if HAVE_TDM
// Issue a 2-D (or flattened 1-D) TENSOR_LOAD_TO_LDS. Whole wave must execute.
// pad_en: insert pad_amount(+1) DWORDs into LDS every 2^(pad_ivl+1) DWORDs.
__device__ __forceinline__ void tdm_load_2d(void* lds_dst, const void* gsrc,
                                            unsigned td0, unsigned td1,
                                            unsigned tile0, unsigned tile1,
                                            unsigned stride0,
                                            int pad_en, unsigned pad_ivl,
                                            unsigned pad_amt)
{
  unsigned long long ga = (unsigned long long)(size_t)gsrc;
  unsigned lds = (unsigned)(size_t)lds_dst;          // LDS aperture: low 32 bits
  v4u g0;
  g0[0] = 1u;                                        // count=1, user descriptor
  g0[1] = lds;                                       // lds_addr
  g0[2] = (unsigned)(ga & 0xFFFFFFFFu);              // global_addr[31:0]
  g0[3] = ((unsigned)(ga >> 32) & 0x01FFFFFFu) | (2u << 30);  // addr[56:32]|type=2
  v8i g1;
  unsigned d1w0 = (2u << 16);                        // data_size=2 (4 bytes)
  if (pad_en) d1w0 |= (1u << 20) | ((pad_ivl & 7u) << 22) | ((pad_amt & 127u) << 25);
  g1[0] = (int)d1w0;
  g1[1] = (int)((td0 & 0xFFFFu) << 16);              // tensor_dim0[15:0]
  g1[2] = (int)(((td0 >> 16) & 0xFFFFu) | ((td1 & 0xFFFFu) << 16));
  g1[3] = (int)(((td1 >> 16) & 0xFFFFu) | ((tile0 & 0xFFFFu) << 16));
  g1[4] = (int)(tile1 & 0xFFFFu);                    // tile_dim1 (tile_dim2=0)
  g1[5] = (int)stride0;                              // tensor_dim0_stride[31:0]
  g1[6] = 0;
  g1[7] = 0;
  v4i gz = {0, 0, 0, 0};
#if defined(__clang_major__) && (__clang_major__ >= 23)
  v8i gz8 = {0, 0, 0, 0, 0, 0, 0, 0};
  __builtin_amdgcn_tensor_load_to_lds(g0, g1, gz, gz, gz8, 0);
#else
  __builtin_amdgcn_tensor_load_to_lds(g0, g1, gz, gz, 0);
#endif
}
#endif

// ---------------------------------------------------------------------------
// Kernel 0: pre-swizzle weights into per-lane B-fragment blocks (bf16).
// Block = 32 lanes x 16 bf16 (1KB). lane: N = nt*16 + lane%16,
// K = kk*32 + (lane/16)*16 + slot   (slot 0..15 contiguous per lane-half).
// ---------------------------------------------------------------------------
__global__ void k_swizzle(const float* __restrict__ eW, const float* __restrict__ aW,
                          const float* __restrict__ sW,
                          unsigned short* __restrict__ swzE,
                          unsigned short* __restrict__ swzA,
                          unsigned short* __restrict__ swzS)
{
  int t = blockIdx.x * blockDim.x + threadIdx.x;
  if (t < 131072) {                      // erase (first 65536) + add
    int which  = t >> 16;
    int e      = t & 65535;
    int blk    = e >> 9;                 // /512
    int within = e & 511;
    int lane   = within >> 4;
    int slot   = within & 15;
    int kk     = blk >> 4;               // NT = 16
    int nt     = blk & 15;
    int n = nt * 16 + (lane & 15);
    int k = kk * 32 + (lane >> 4) * 16 + slot;
    const float* W = which ? aW : eW;
    unsigned short* dst = which ? swzA : swzE;
    dst[e] = f2bf(W[k * 256 + n]);
  } else if (t < 151552) {               // summary
    int e      = t - 131072;             // [0, 20480)
    int blk    = e >> 9;
    int within = e & 511;
    int lane   = within >> 4;
    int slot   = within & 15;
    int kk     = blk >> 2;               // NT = 4
    int nt     = blk & 3;
    int n = nt * 16 + (lane & 15);
    int k = kk * 32 + (lane >> 4) * 16 + slot;
    float val = (n < 50) ? sW[k * 50 + n] : 0.0f;
    swzS[e] = f2bf(val);
  }
}

// ---------------------------------------------------------------------------
// Kernel 1: per 32-row tile: qe gather, attention softmax w, ve gather (bf16
// LDS A-tile), then fused erase/add GEMM with WMMA + activation epilogues.
// 256 threads = 8 waves; waves 0-3 -> erase cols, waves 4-7 -> add cols.
// Mk staged into padded LDS by the Tensor Data Mover (pad 4 dwords / 64).
// ---------------------------------------------------------------------------
__global__ __launch_bounds__(256) void k_prep_gemm(
    const int* __restrict__ q_data, const int* __restrict__ r_data,
    const float* __restrict__ qeW, const float* __restrict__ MkW,
    const float* __restrict__ vW, const float* __restrict__ vb,
    const float* __restrict__ ebias, const float* __restrict__ abias,
    const unsigned short* __restrict__ swzE, const unsigned short* __restrict__ swzA,
    float* __restrict__ w_out, float* __restrict__ eact, float* __restrict__ aact)
{
  __shared__ float qe[32][64];
  __shared__ float mk[50][68];
  __shared__ float sc[32][52];
  __shared__ __align__(16) unsigned short At[32][264];  // ve tile, bf16, padded
  __shared__ int   s_q[32];
  __shared__ int   s_idx[32];
  __shared__ int   s_valid[32];
  __shared__ float s_rs[32];
  __shared__ float s_inv[32];

  const int tid = threadIdx.x;
  const long base = (long)blockIdx.x * 32;

  if (tid < 32) {
    int q = q_data[base + tid];
    int r = r_data[base + tid];
    s_q[tid]     = q;
    s_idx[tid]   = (q > 0) ? (q - 1) : 0;
    s_valid[tid] = (q > 0) ? 1 : 0;
    s_rs[tid]    = (float)r * 0.25f;     // r / (K-1)
  }
#if HAVE_TDM
  if ((tid >> 5) == 0) {                 // wave 0 issues the Mk DMA
    // 50x64 fp32 rows; TDM pads 4 dwords after every 64 -> LDS stride 68.
    tdm_load_2d(&mk[0][0], MkW, /*td0=*/64, /*td1=*/50, /*tile0=*/64,
                /*tile1=*/50, /*stride0=*/64, /*pad_en=*/1,
                /*pad_ivl=*/5, /*pad_amt=*/3);
    __builtin_amdgcn_s_wait_tensorcnt(0);
  }
#else
  for (int e = tid; e < 50 * 64; e += 256) {
    int m = e >> 6, c = e & 63;
    mk[m][c] = MkW[m * 64 + c];
  }
#endif
  __syncthreads();

  for (int e = tid; e < 2048; e += 256) {          // qe gather
    int row = e >> 6, c = e & 63;
    qe[row][c] = qeW[s_q[row] * 64 + c];
  }
  for (int e = tid; e < 8192; e += 256) {          // ve gather -> bf16
    int row = e >> 8, c = e & 255;
    float v = 0.0f;
    if (s_valid[row]) {
      int ix = s_idx[row];
      v = vW[(size_t)ix * 256 + c] + s_rs[row] * vW[(size_t)(NQn + ix) * 256 + c];
    }
    v += vb[c];
    At[row][c] = f2bf(v);
  }
  __syncthreads();

  for (int e = tid; e < 1600; e += 256) {          // scores qe . Mk[m]
    int row = e / 50, m = e - row * 50;
    float s = 0.0f;
    for (int c = 0; c < 64; ++c) s += qe[row][c] * mk[m][c];
    sc[row][m] = s;
  }
  __syncthreads();

  if (tid < 32) {                                  // softmax (per row)
    float mx = sc[tid][0];
    for (int m = 1; m < 50; ++m) mx = fmaxf(mx, sc[tid][m]);
    float sum = 0.0f;
    for (int m = 0; m < 50; ++m) { float ev = __expf(sc[tid][m] - mx); sc[tid][m] = ev; sum += ev; }
    s_inv[tid] = 1.0f / sum;
  }
  __syncthreads();
  for (int e = tid; e < 1600; e += 256) {
    int row = e / 50, m = e - row * 50;
    w_out[(size_t)(base + row) * 64 + m] = sc[row][m] * s_inv[row];
  }

  // ---- WMMA GEMM: sigmoid(ve@eW+eb) and tanh(ve@aW+ab) ----
  const int lane = tid & 31;
  const int wv   = tid >> 5;
  const int mat  = wv >> 2;               // 0 = erase, 1 = add
  const int nt0  = (wv & 3) * 4;
  const unsigned short* swz = mat ? swzA : swzE;
  const float* bias = mat ? abias : ebias;
  float* outp = mat ? aact : eact;
  const v16bf* bsw = (const v16bf*)(const void*)swz;
  const int m16 = lane & 15;
  const int lhalf = lane >> 4;

  for (int rb = 0; rb < 2; ++rb) {
    const unsigned int* arow = (const unsigned int*)(const void*)&At[rb * 16 + m16][0];
    for (int ntl = 0; ntl < 4; ++ntl) {
      int nt = nt0 + ntl;
      v8f acc;
      #pragma unroll
      for (int i = 0; i < 8; ++i) acc[i] = 0.0f;
      for (int kk = 0; kk < 8; ++kk) {
        Frag16 af;
        #pragma unroll
        for (int vv = 0; vv < 8; ++vv) {   // 16-bit A layout (16x32)
          int k = kk * 32 + lhalf * 16 + vv * 2;
          af.u[vv] = arow[k >> 1];
        }
        v16bf bfrag = bsw[(kk * 16 + nt) * 32 + lane];
        acc = __builtin_amdgcn_wmma_f32_16x16x32_bf16(false, af.v, false, bfrag,
                                                      (short)0, acc, false, false);
      }
      int col = nt * 16 + m16;
      float bcol = bias[col];
      #pragma unroll
      for (int i = 0; i < 8; ++i) {        // C layout: M = i + 8*(lane>=16)
        int rl = rb * 16 + i + lhalf * 8;
        float x = acc[i] + bcol;
        float rres = (mat == 0) ? sigmoidf_(x) : tanhf(x);
        outp[(size_t)(base + rl) * 256 + col] = rres;
      }
    }
  }
}

// ---------------------------------------------------------------------------
// Kernel 2: sequential scan over T. One workgroup per batch element; state
// Mv[50][256] lives in LDS (initialized via TDM); each thread owns one DV
// column (fused read + rank-1 update preserves read-before-write semantics).
// ---------------------------------------------------------------------------
__global__ __launch_bounds__(256) void k_scan(
    const float* __restrict__ Mv0, const float* __restrict__ w_ws,
    const float* __restrict__ eact, const float* __restrict__ aact,
    float* __restrict__ reads)
{
  __shared__ float Mv[50 * 256];
  __shared__ float wb[64];
  const int tid = threadIdx.x;
  const long rowbase = (long)blockIdx.x * 512;

#if HAVE_TDM
  if ((tid >> 5) == 0) {                 // wave 0: DMA the whole 51.2KB state
    tdm_load_2d(&Mv[0], Mv0, /*td0=*/12800, /*td1=*/1, /*tile0=*/12800,
                /*tile1=*/1, /*stride0=*/12800, /*pad_en=*/0, 0, 0);
    __builtin_amdgcn_s_wait_tensorcnt(0);
  }
#else
  for (int e = tid; e < 12800; e += 256) Mv[e] = Mv0[e];
#endif
  __syncthreads();

  for (int t = 0; t < 512; ++t) {
    const long rt = rowbase + t;
    if (tid < 50) wb[tid] = w_ws[rt * 64 + tid];
    __syncthreads();
    float ev = eact[rt * 256 + tid];
    float av = aact[rt * 256 + tid];
    if (t + 1 < 512) {                      // global_prefetch_b8 for next step
      __builtin_prefetch(eact + (rt + 1) * 256 + tid, 0, 1);
      __builtin_prefetch(aact + (rt + 1) * 256 + tid, 0, 1);
    }
    float rd = 0.0f;
    #pragma unroll 10
    for (int m = 0; m < 50; ++m) {
      float wm  = wb[m];
      float old = Mv[m * 256 + tid];
      rd += wm * old;
      Mv[m * 256 + tid] = old * (1.0f - wm * ev) + wm * av;
    }
    reads[rt * 256 + tid] = rd;
    __syncthreads();
  }
}

// ---------------------------------------------------------------------------
// Kernel 3: summary = tanh([reads,qe] @ summary_W + b) via WMMA (K=320,
// N padded to 64), then all scalar heads (theta/betas/alpha/logits/probs).
// ---------------------------------------------------------------------------
__global__ __launch_bounds__(256) void k_head(
    const int* __restrict__ q_data, const float* __restrict__ qeW,
    const float* __restrict__ reads, const unsigned short* __restrict__ swzS,
    const float* __restrict__ sb,
    const float* __restrict__ abilW, const float* __restrict__ abilB,
    const float* __restrict__ thW, const float* __restrict__ thB,
    const float* __restrict__ dW, const float* __restrict__ dB,
    const float* __restrict__ cW, const float* __restrict__ cB,
    float* __restrict__ out)
{
  __shared__ float qe[32][65];
  __shared__ __align__(16) unsigned short At[32][328];   // [reads(256) | qe(64)]
  __shared__ float sm[32][68];
  const int tid = threadIdx.x;
  const long base = (long)blockIdx.x * 32;

  for (int e = tid; e < 2048; e += 256) {
    int row = e >> 6, c = e & 63;
    float v = qeW[q_data[base + row] * 64 + c];
    qe[row][c] = v;
    At[row][256 + c] = f2bf(v);
  }
  for (int e = tid; e < 8192; e += 256) {
    int row = e >> 8, c = e & 255;
    At[row][c] = f2bf(reads[(size_t)(base + row) * 256 + c]);
  }
  __syncthreads();

  const int lane = tid & 31, wv = tid >> 5;
  const int rb = wv >> 2, nt = wv & 3;       // 8 waves cover 2 rb x 4 nt tiles
  const int m16 = lane & 15, lhalf = lane >> 4;
  const v16bf* bsw = (const v16bf*)(const void*)swzS;
  const unsigned int* arow = (const unsigned int*)(const void*)&At[rb * 16 + m16][0];
  v8f acc;
  #pragma unroll
  for (int i = 0; i < 8; ++i) acc[i] = 0.0f;
  for (int kk = 0; kk < 10; ++kk) {
    Frag16 af;
    #pragma unroll
    for (int vv = 0; vv < 8; ++vv) {
      int k = kk * 32 + lhalf * 16 + vv * 2;
      af.u[vv] = arow[k >> 1];
    }
    v16bf bfrag = bsw[(kk * 4 + nt) * 32 + lane];
    acc = __builtin_amdgcn_wmma_f32_16x16x32_bf16(false, af.v, false, bfrag,
                                                  (short)0, acc, false, false);
  }
  {
    int col = nt * 16 + m16;
    float bcol = (col < 50) ? sb[col] : 0.0f;
    #pragma unroll
    for (int i = 0; i < 8; ++i) {
      int rl = rb * 16 + i + lhalf * 8;
      sm[rl][col] = tanhf(acc[i] + bcol);
    }
  }
  __syncthreads();

  if (tid < 32) {
    const long bt = base + tid;
    float th = abilB[0];
    for (int c = 0; c < 50; ++c) th += sm[tid][c] * abilW[c];
    out[bt] = th * 3.0f;                                    // theta
    for (int k = 0; k < 4; ++k) {                           // betas
      float v = thB[k];
      for (int c = 0; c < 64; ++c) v += qe[tid][c] * thW[c * 4 + k];
      out[NBT + bt * 4 + k] = v;
    }
    float dv = dB[0];                                       // alpha
    for (int c = 0; c < 50; ++c) dv += sm[tid][c] * dW[c];
    for (int c = 0; c < 64; ++c) dv += qe[tid][c] * dW[50 + c];
    out[5 * NBT + bt] = (dv > 20.0f) ? dv : log1pf(__expf(dv));
    float cp[4]; float run = 1.0f;
    for (int k = 0; k < 4; ++k) {                           // logits + cumprod
      float v = cB[k];
      for (int c = 0; c < 50; ++c) v += sm[tid][c] * cW[c * 4 + k];
      for (int c = 0; c < 64; ++c) v += qe[tid][c] * cW[(50 + c) * 4 + k];
      out[11 * NBT + bt * 4 + k] = v;
      run *= sigmoidf_(v);
      cp[k] = run;
    }
    out[6 * NBT + bt * 5 + 0] = 1.0f - cp[0];               // CORN probs
    out[6 * NBT + bt * 5 + 1] = cp[0] - cp[1];
    out[6 * NBT + bt * 5 + 2] = cp[1] - cp[2];
    out[6 * NBT + bt * 5 + 3] = cp[2] - cp[3];
    out[6 * NBT + bt * 5 + 4] = cp[3];
  }
}

// ---------------------------------------------------------------------------
extern "C" void kernel_launch(void* const* d_in, const int* in_sizes, int n_in,
                              void* d_out, int out_size, void* d_ws, size_t ws_size,
                              hipStream_t stream)
{
  const int*   q_data = (const int*)  d_in[0];
  const int*   r_data = (const int*)  d_in[1];
  const float* qeW    = (const float*)d_in[2];
  const float* MkW    = (const float*)d_in[3];
  const float* Mv0    = (const float*)d_in[4];
  const float* vW     = (const float*)d_in[5];
  const float* vb     = (const float*)d_in[6];
  const float* eW     = (const float*)d_in[7];
  const float* ebias  = (const float*)d_in[8];
  const float* aW     = (const float*)d_in[9];
  const float* abias  = (const float*)d_in[10];
  const float* sW     = (const float*)d_in[11];
  const float* sb     = (const float*)d_in[12];
  const float* abilW  = (const float*)d_in[13];
  const float* abilB  = (const float*)d_in[14];
  const float* thW    = (const float*)d_in[15];
  const float* thB    = (const float*)d_in[16];
  const float* dW     = (const float*)d_in[17];
  const float* dB     = (const float*)d_in[18];
  const float* cW     = (const float*)d_in[19];
  const float* cB     = (const float*)d_in[20];

  char* ws = (char*)d_ws;
  unsigned short* swzE = (unsigned short*)ws;                 ws += 131072;
  unsigned short* swzA = (unsigned short*)ws;                 ws += 131072;
  unsigned short* swzS = (unsigned short*)ws;                 ws += 65536;
  float* w_ws  = (float*)ws;  ws += (size_t)NBT * 64  * 4;
  float* eact  = (float*)ws;  ws += (size_t)NBT * 256 * 4;
  float* aact  = (float*)ws;  ws += (size_t)NBT * 256 * 4;
  float* reads = (float*)ws;  ws += (size_t)NBT * 256 * 4;

  k_swizzle<<<592, 256, 0, stream>>>(eW, aW, sW, swzE, swzA, swzS);
  k_prep_gemm<<<8192, 256, 0, stream>>>(q_data, r_data, qeW, MkW, vW, vb,
                                        ebias, abias, swzE, swzA,
                                        w_ws, eact, aact);
  k_scan<<<512, 256, 0, stream>>>(Mv0, w_ws, eact, aact, reads);
  k_head<<<8192, 256, 0, stream>>>(q_data, qeW, reads, swzS, sb,
                                   abilW, abilB, thW, thB, dW, dB, cW, cB,
                                   (float*)d_out);
}
